// AtomGraphGINE_40750649704710
// MI455X (gfx1250) — matrix operand order repaired and checked by reading, not compiled, not run on hardware
//
#include <hip/hip_runtime.h>

#define N_NODES   100000
#define N_EDGES   400000
#define N_GRAPHS  4096
#define ATOM_DIM  101
#define BOND_DIM  11
#define D_LAT     128
#define D_HID     256
#define N_LAYERS  3
#define BN_EPS    1e-5f
#define MBLOCK    5                      // m-tiles per wave (6250 = 1250*5)
#define MTILES    (N_NODES / 16)         // 6250
#define MBLOCKS   (MTILES / MBLOCK)      // 1250

typedef __attribute__((ext_vector_type(16))) __bf16 v16bf;
typedef __attribute__((ext_vector_type(8)))  float  v8f;

// v16bf viewed as 8 packed dwords (each = two bf16, low half = even element)
union FragU {
  v16bf v;
  unsigned u[8];
};

__device__ __forceinline__ unsigned pk_bf16(float a, float b) {
  unsigned short lo = __builtin_bit_cast(unsigned short, (__bf16)a);
  unsigned short hi = __builtin_bit_cast(unsigned short, (__bf16)b);
  return ((unsigned)hi << 16) | (unsigned)lo;   // -> v_cvt_pk_bf16_f32
}

__device__ __forceinline__ void atomAddF(float* p, float v) {
  __hip_atomic_fetch_add(p, v, __ATOMIC_RELAXED, __HIP_MEMORY_SCOPE_AGENT);
}

// K offset inside a 16x32 bf16 A fragment for dword j (pair j), per ISA 7.12.2
__device__ __forceinline__ int a_koff(int j, int hi) {
  return ((j < 4) ? 0 : 16) + hi * 8 + (j & 3) * 2;
}

// ---------------------------------------------------------------------------
// Fragment loaders (wave32 layouts per CDNA5 ISA 7.12.2)
// ---------------------------------------------------------------------------

// A fragment from pre-converted bf16 row-major activations [.. ][K] (K-contig)
__device__ __forceinline__ v16bf load_A_bf16(const __bf16* __restrict__ xb,
                                             int K, int m0, int k0, int lane) {
  int m  = m0 + (lane & 15);
  int hi = lane >> 4;
  const __bf16* row = xb + (size_t)m * K + k0;
  FragU A;
#pragma unroll
  for (int j = 0; j < 8; ++j)
    A.u[j] = *(const unsigned*)(row + a_koff(j, hi));
  return A.v;
}

// A fragment (16x32 bf16) from z = h + agg, rows stride 128 f32
__device__ __forceinline__ v16bf load_A_zsum(const float* __restrict__ h,
                                             const float* __restrict__ agg,
                                             int m0, int k0, int lane) {
  int m  = m0 + (lane & 15);
  int hi = lane >> 4;
  const float* hr = h   + (size_t)m * D_LAT + k0;
  const float* ar = agg + (size_t)m * D_LAT + k0;
  FragU A;
#pragma unroll
  for (int j = 0; j < 8; ++j) {
    int k = a_koff(j, hi);
    float2 hz = *(const float2*)(hr + k);
    float2 az = *(const float2*)(ar + k);
    A.u[j] = pk_bf16(hz.x + az.x, hz.y + az.y);
  }
  return A.v;
}

// A fragment from t with fused BatchNorm + ReLU (per-column scale/shift)
__device__ __forceinline__ v16bf load_A_bnrelu(const float* __restrict__ t,
                                               const float* __restrict__ scale,
                                               const float* __restrict__ shift,
                                               int m0, int k0, int lane) {
  int m  = m0 + (lane & 15);
  int hi = lane >> 4;
  const float* row = t + (size_t)m * D_HID + k0;
  FragU A;
#pragma unroll
  for (int j = 0; j < 8; ++j) {
    int k = a_koff(j, hi);
    float2 tv = *(const float2*)(row + k);
    float2 sc = *(const float2*)(scale + k0 + k);
    float2 sh = *(const float2*)(shift + k0 + k);
    float v0 = fmaxf(fmaf(tv.x, sc.x, sh.x), 0.f);
    float v1 = fmaxf(fmaf(tv.y, sc.y, sh.y), 0.f);
    A.u[j] = pk_bf16(v0, v1);
  }
  return A.v;
}

// B fragment (32x16 bf16) from transposed weights WT[n][K] (K-contiguous)
__device__ __forceinline__ v16bf load_B_T(const __bf16* __restrict__ WT,
                                          int K, int n0, int k0, int lane) {
  int n  = n0 + (lane & 15);
  int hi = lane >> 4;
  const unsigned* col =
      (const unsigned*)(WT + (size_t)n * K + k0);
  FragU B;
#pragma unroll
  for (int j = 0; j < 8; ++j) B.u[j] = col[a_koff(j, hi) >> 1];
  return B.v;
}

// ---------------------------------------------------------------------------
// Utility / conversion kernels
// ---------------------------------------------------------------------------

__global__ void zero_f32(float* __restrict__ p, long long n) {
  long long i = (long long)blockIdx.x * blockDim.x + threadIdx.x;
  if (i < n) p[i] = 0.f;
}

// x [N][101] f32 -> xb bf16 [N][128], zero-padded K
__global__ void conv_xpad(const float* __restrict__ x, __bf16* __restrict__ xb) {
  long long i = (long long)blockIdx.x * blockDim.x + threadIdx.x;  // N*128
  if (i < (long long)N_NODES * 128) {
    int n = (int)(i >> 7), k = (int)(i & 127);
    xb[i] = (k < ATOM_DIM) ? (__bf16)x[(size_t)n * ATOM_DIM + k] : (__bf16)0.f;
  }
}

// embW [101][128] -> embWT bf16 [128 cols][128 K], zero-padded past K=101
__global__ void conv_embW(const float* __restrict__ W, __bf16* __restrict__ WT) {
  int i = blockIdx.x * blockDim.x + threadIdx.x;  // 128*128
  int n = i >> 7, k = i & 127;
  WT[n * 128 + k] = (k < ATOM_DIM) ? (__bf16)W[k * D_LAT + n] : (__bf16)0.f;
}

// W1 [L][128][256] -> W1T bf16 [L][256][128]
__global__ void conv_W1(const float* __restrict__ W, __bf16* __restrict__ WT) {
  int i = blockIdx.x * blockDim.x + threadIdx.x;  // 3*256*128
  int l = i / (D_HID * D_LAT);
  int r = i % (D_HID * D_LAT);
  int n = r >> 7, k = r & 127;
  WT[i] = (__bf16)W[(size_t)l * D_LAT * D_HID + (size_t)k * D_HID + n];
}

// W2 [L][256][128] -> W2T bf16 [L][128][256]
__global__ void conv_W2(const float* __restrict__ W, __bf16* __restrict__ WT) {
  int i = blockIdx.x * blockDim.x + threadIdx.x;  // 3*128*256
  int l = i / (D_LAT * D_HID);
  int r = i % (D_LAT * D_HID);
  int n = r >> 8, k = r & 255;
  WT[i] = (__bf16)W[(size_t)l * D_HID * D_LAT + (size_t)k * D_LAT + n];
}

// ---------------------------------------------------------------------------
// GEMM kernels: one n-tile per wave, B fragments register-resident across
// MBLOCK m-tiles; 8 waves / 256-thread block.
// ---------------------------------------------------------------------------

// h = x @ embW + embB   (K padded to 128, x pre-converted bf16)
__global__ void embed_gemm(const __bf16* __restrict__ xb,
                           const __bf16* __restrict__ embWT,
                           const float* __restrict__ embB,
                           float* __restrict__ h) {
  int wave = (blockIdx.x * blockDim.x + threadIdx.x) >> 5;
  int lane = threadIdx.x & 31;
  int mblk = wave >> 3, ntile = wave & 7;       // 8 n-tiles of 16
  int n0 = ntile * 16;
  v16bf B[4];
#pragma unroll
  for (int kk = 0; kk < 4; ++kk) B[kk] = load_B_T(embWT, 128, n0, kk * 32, lane);
  int n = n0 + (lane & 15), hi = lane >> 4;
  float bias = embB[n];
  for (int mi = 0; mi < MBLOCK; ++mi) {
    int m0 = (mblk * MBLOCK + mi) * 16;
    v8f C = {};
#pragma unroll
    for (int kk = 0; kk < 4; ++kk) {
      v16bf A = load_A_bf16(xb, 128, m0, kk * 32, lane);
      C = __builtin_amdgcn_wmma_f32_16x16x32_bf16(false, A, false, B[kk],
                                                  (short)0, C, false, false);
    }
#pragma unroll
    for (int r = 0; r < 8; ++r)
      h[(size_t)(m0 + 8 * hi + r) * D_LAT + n] = C[r] + bias;
  }
}

// t = (h + agg) @ W1 + b1   [N,128]@[128,256] ; accumulates column sum/sumsq
__global__ void gemm1_kernel(const float* __restrict__ h,
                             const float* __restrict__ agg,
                             const __bf16* __restrict__ W1T,
                             const float* __restrict__ b1,
                             float* __restrict__ t,
                             float* __restrict__ sum1,
                             float* __restrict__ sumsq1) {
  int wave = (blockIdx.x * blockDim.x + threadIdx.x) >> 5;
  int lane = threadIdx.x & 31;
  int mblk = wave >> 4, ntile = wave & 15;      // 16 n-tiles of 16
  int n0 = ntile * 16;
  v16bf B[4];
#pragma unroll
  for (int kk = 0; kk < 4; ++kk) B[kk] = load_B_T(W1T, D_LAT, n0, kk * 32, lane);
  int n = n0 + (lane & 15), hi = lane >> 4;
  float bias = b1[n];
  float s = 0.f, ss = 0.f;
  for (int mi = 0; mi < MBLOCK; ++mi) {
    int m0 = (mblk * MBLOCK + mi) * 16;
    // prefetch next m-tile's activation rows while WMMAs run
    __builtin_prefetch(h + (size_t)(m0 + 16 + (lane & 15)) * D_LAT, 0, 1);
    v8f C = {};
#pragma unroll
    for (int kk = 0; kk < 4; ++kk) {
      v16bf A = load_A_zsum(h, agg, m0, kk * 32, lane);
      C = __builtin_amdgcn_wmma_f32_16x16x32_bf16(false, A, false, B[kk],
                                                  (short)0, C, false, false);
    }
#pragma unroll
    for (int r = 0; r < 8; ++r) {
      float v = C[r] + bias;
      t[(size_t)(m0 + 8 * hi + r) * D_HID + n] = v;
      s += v;
      ss += v * v;
    }
  }
  atomAddF(&sum1[n], s);
  atomAddF(&sumsq1[n], ss);
}

// u = relu(bn(t)) @ W2 + b2   [N,256]@[256,128] ; accumulates column sum/sumsq
__global__ void gemm2_kernel(const float* __restrict__ t,
                             const float* __restrict__ scale1,
                             const float* __restrict__ shift1,
                             const __bf16* __restrict__ W2T,
                             const float* __restrict__ b2,
                             float* __restrict__ u,
                             float* __restrict__ sum2,
                             float* __restrict__ sumsq2) {
  int wave = (blockIdx.x * blockDim.x + threadIdx.x) >> 5;
  int lane = threadIdx.x & 31;
  int mblk = wave >> 3, ntile = wave & 7;       // 8 n-tiles of 16
  int n0 = ntile * 16;
  v16bf B[8];
#pragma unroll
  for (int kk = 0; kk < 8; ++kk) B[kk] = load_B_T(W2T, D_HID, n0, kk * 32, lane);
  int n = n0 + (lane & 15), hi = lane >> 4;
  float bias = b2[n];
  float s = 0.f, ss = 0.f;
  for (int mi = 0; mi < MBLOCK; ++mi) {
    int m0 = (mblk * MBLOCK + mi) * 16;
    __builtin_prefetch(t + (size_t)(m0 + 16 + (lane & 15)) * D_HID, 0, 1);
    v8f C = {};
#pragma unroll
    for (int kk = 0; kk < 8; ++kk) {
      v16bf A = load_A_bnrelu(t, scale1, shift1, m0, kk * 32, lane);
      C = __builtin_amdgcn_wmma_f32_16x16x32_bf16(false, A, false, B[kk],
                                                  (short)0, C, false, false);
    }
#pragma unroll
    for (int r = 0; r < 8; ++r) {
      float v = C[r] + bias;
      u[(size_t)(m0 + 8 * hi + r) * D_LAT + n] = v;
      s += v;
      ss += v * v;
    }
  }
  atomAddF(&sum2[n], s);
  atomAddF(&sumsq2[n], ss);
}

// ---------------------------------------------------------------------------
// BatchNorm stats fold: scale = gamma*rsqrt(var+eps), shift = beta - mean*scale
// ---------------------------------------------------------------------------
__global__ void finalize_stats(const float* __restrict__ sum,
                               const float* __restrict__ sumsq,
                               const float* __restrict__ gamma,
                               const float* __restrict__ beta,
                               float* __restrict__ scale,
                               float* __restrict__ shift,
                               int C, float invN) {
  int c = blockIdx.x * blockDim.x + threadIdx.x;
  if (c < C) {
    float m   = sum[c] * invN;
    float var = sumsq[c] * invN - m * m;
    float sc  = rsqrtf(var + BN_EPS) * gamma[c];
    scale[c] = sc;
    shift[c] = beta[c] - m * sc;
  }
}

// h = bn(u) (+relu)  elementwise over N*128
__global__ void apply_bn(const float* __restrict__ u,
                         const float* __restrict__ scale,
                         const float* __restrict__ shift,
                         float* __restrict__ h, int relu) {
  long long i = (long long)blockIdx.x * blockDim.x + threadIdx.x;
  if (i < (long long)N_NODES * D_LAT) {
    int f = (int)(i & (D_LAT - 1));
    float v = fmaf(u[i], scale[f], shift[f]);
    if (relu) v = fmaxf(v, 0.f);
    h[i] = v;
  }
}

// ---------------------------------------------------------------------------
// Edge kernel: e = edge_attr @ bondW + bondB; msg = relu(h[src]+e);
// agg[dst] += msg   (2 edges / 256-thread block, 128 features each)
// ---------------------------------------------------------------------------
__global__ void edge_msg(const float* __restrict__ h,
                         const float* __restrict__ edge_attr,
                         const float* __restrict__ bondW,   // layer slice [11][128]
                         const float* __restrict__ bondB,   // layer slice [128]
                         const int* __restrict__ src,
                         const int* __restrict__ dst,
                         float* __restrict__ agg) {
  __shared__ float ea[2][BOND_DIM];
  __shared__ int   sd[2][2];
  int f  = threadIdx.x & 127;
  int el = threadIdx.x >> 7;
  int e  = blockIdx.x * 2 + el;
  if (f < BOND_DIM) ea[el][f] = edge_attr[(size_t)e * BOND_DIM + f];
  if (f == 0) { sd[el][0] = src[e]; sd[el][1] = dst[e]; }
  __syncthreads();
  float acc = bondB[f];
#pragma unroll
  for (int k = 0; k < BOND_DIM; ++k)
    acc = fmaf(ea[el][k], bondW[k * D_LAT + f], acc);
  float m = fmaxf(h[(size_t)sd[el][0] * D_LAT + f] + acc, 0.f);
  atomAddF(&agg[(size_t)sd[el][1] * D_LAT + f], m);
}

// ---------------------------------------------------------------------------
// Global mean pool
// ---------------------------------------------------------------------------
__global__ void pool_accum(const float* __restrict__ h,
                           const int* __restrict__ batch,
                           float* __restrict__ out,
                           float* __restrict__ cnt) {
  int n = blockIdx.x;
  int f = threadIdx.x;
  int g = batch[n];
  atomAddF(&out[(size_t)g * D_LAT + f], h[(size_t)n * D_LAT + f]);
  if (f == 0) atomAddF(&cnt[g], 1.0f);
}

__global__ void pool_div(float* __restrict__ out, const float* __restrict__ cnt) {
  int i = blockIdx.x * blockDim.x + threadIdx.x;
  if (i < N_GRAPHS * D_LAT) out[i] /= fmaxf(cnt[i >> 7], 1.0f);
}

// ---------------------------------------------------------------------------
// Launch
// ---------------------------------------------------------------------------
extern "C" void kernel_launch(void* const* d_in, const int* in_sizes, int n_in,
                              void* d_out, int out_size, void* d_ws, size_t ws_size,
                              hipStream_t stream) {
  (void)in_sizes; (void)n_in; (void)out_size; (void)ws_size;

  const float* x          = (const float*)d_in[0];
  const float* edge_attr  = (const float*)d_in[1];
  const float* embW       = (const float*)d_in[2];
  const float* embB       = (const float*)d_in[3];
  const float* bondW      = (const float*)d_in[4];
  const float* bondB      = (const float*)d_in[5];
  const float* W1         = (const float*)d_in[6];
  const float* b1         = (const float*)d_in[7];
  const float* g1         = (const float*)d_in[8];
  const float* be1        = (const float*)d_in[9];
  const float* W2         = (const float*)d_in[10];
  const float* b2         = (const float*)d_in[11];
  const float* gout       = (const float*)d_in[12];
  const float* bout       = (const float*)d_in[13];
  const int*   edge_index = (const int*)d_in[14];
  const int*   batch      = (const int*)d_in[15];
  float* out = (float*)d_out;

  // workspace carve-up
  char* ws = (char*)d_ws;
  size_t off = 0;
  auto take = [&](size_t bytes) -> char* {
    char* p = ws + off;
    off += (bytes + 255) & ~(size_t)255;
    return p;
  };
  float*  h      = (float*)take((size_t)N_NODES * D_LAT * 4);
  float*  agg    = (float*)take((size_t)N_NODES * D_LAT * 4);  // reused as u
  float*  t      = (float*)take((size_t)N_NODES * D_HID * 4);
  __bf16* xb     = (__bf16*)take((size_t)N_NODES * 128 * 2);
  __bf16* embWT  = (__bf16*)take((size_t)128 * 128 * 2);
  __bf16* W1T    = (__bf16*)take((size_t)N_LAYERS * D_HID * D_LAT * 2);
  __bf16* W2T    = (__bf16*)take((size_t)N_LAYERS * D_LAT * D_HID * 2);
  float*  sum1   = (float*)take(256 * 4);   // sum1|sumsq1|sum2|sumsq2 contiguous
  float*  sumsq1 = (float*)take(256 * 4);
  float*  sum2   = (float*)take(128 * 4);
  float*  sumsq2 = (float*)take(128 * 4);
  float*  scale1 = (float*)take(256 * 4);
  float*  shift1 = (float*)take(256 * 4);
  float*  scale2 = (float*)take(128 * 4);
  float*  shift2 = (float*)take(128 * 4);
  float*  cnt    = (float*)take((size_t)N_GRAPHS * 4);

  const int* src = edge_index;
  const int* dst = edge_index + N_EDGES;
  const float invN = 1.0f / (float)N_NODES;

  // pre-convert activations/weights to bf16 (transposed, fragment-friendly)
  conv_xpad<<<((long long)N_NODES * 128 + 255) / 256, 256, 0, stream>>>(x, xb);
  conv_embW<<<(128 * 128) / 256, 256, 0, stream>>>(embW, embWT);
  conv_W1<<<(N_LAYERS * D_HID * D_LAT) / 256, 256, 0, stream>>>(W1, W1T);
  conv_W2<<<(N_LAYERS * D_LAT * D_HID) / 256, 256, 0, stream>>>(W2, W2T);

  // h = x @ embW + embB : 1250 m-blocks * 8 n-tiles = 10000 waves
  embed_gemm<<<(MBLOCKS * 8) / 8, 256, 0, stream>>>(xb, embWT, embB, h);

  for (int i = 0; i < N_LAYERS; ++i) {
    zero_f32<<<((long long)N_NODES * D_LAT + 255) / 256, 256, 0, stream>>>(
        agg, (long long)N_NODES * D_LAT);
    zero_f32<<<3, 256, 0, stream>>>(sum1, 768);  // all four stat buffers

    edge_msg<<<N_EDGES / 2, 256, 0, stream>>>(
        h, edge_attr, bondW + (size_t)i * BOND_DIM * D_LAT,
        bondB + (size_t)i * D_LAT, src, dst, agg);

    gemm1_kernel<<<(MBLOCKS * 16) / 8, 256, 0, stream>>>(
        h, agg, W1T + (size_t)i * D_HID * D_LAT, b1 + (size_t)i * D_HID,
        t, sum1, sumsq1);

    finalize_stats<<<1, 256, 0, stream>>>(sum1, sumsq1,
                                          g1 + (size_t)i * D_HID,
                                          be1 + (size_t)i * D_HID,
                                          scale1, shift1, D_HID, invN);

    // u written into agg (dead after gemm1)
    gemm2_kernel<<<(MBLOCKS * 8) / 8, 256, 0, stream>>>(
        t, scale1, shift1, W2T + (size_t)i * D_LAT * D_HID,
        b2 + (size_t)i * D_LAT, agg, sum2, sumsq2);

    finalize_stats<<<1, 128, 0, stream>>>(sum2, sumsq2,
                                          gout + (size_t)i * D_LAT,
                                          bout + (size_t)i * D_LAT,
                                          scale2, shift2, D_LAT, invN);

    apply_bn<<<((long long)N_NODES * D_LAT + 255) / 256, 256, 0, stream>>>(
        agg, scale2, shift2, h, (i != N_LAYERS - 1) ? 1 : 0);
  }

  // global mean pool
  zero_f32<<<(N_GRAPHS * D_LAT + 255) / 256, 256, 0, stream>>>(
      out, (long long)N_GRAPHS * D_LAT);
  zero_f32<<<(N_GRAPHS + 255) / 256, 256, 0, stream>>>(cnt, N_GRAPHS);
  pool_accum<<<N_NODES, D_LAT, 0, stream>>>(h, batch, out, cnt);
  pool_div<<<(N_GRAPHS * D_LAT) / 256, 256, 0, stream>>>(out, cnt);
}
